// Decoder_15960098472131
// MI455X (gfx1250) — matrix operand order repaired and checked by reading, not compiled, not run on hardware
//
#include <hip/hip_runtime.h>

// Problem constants (match reference)
#define BB 4
#define TT 2048
#define EE 1024
#define HH 16
#define HD 64
#define MTOK (BB * TT)   // 8192 tokens

typedef __attribute__((ext_vector_type(16))) __bf16 v16bf;
typedef __attribute__((ext_vector_type(8)))  float  v8f;

__device__ __forceinline__ unsigned short f2bf(float f) {
  unsigned int u = __float_as_uint(f);
  // round-to-nearest-even bf16
  return (unsigned short)((u + 0x7FFFu + ((u >> 16) & 1u)) >> 16);
}

__device__ __forceinline__ v8f zero_v8f() {
  v8f z;
#pragma unroll
  for (int i = 0; i < 8; ++i) z[i] = 0.0f;
  return z;
}

union FragBF { unsigned int u[8]; v16bf v; };

// A fragment: 16x32 bf16. Source row-major, pre-offset to (rbase, kbase), leading dim lda.
// ISA 7.12.2 16-bit A 16x32: lanes 0-15 rows 0-15 hold K {0..7,16..23},
// lanes 16-31 hold K {8..15,24..31}; VGPR v packs a K pair -> one dword load.
__device__ __forceinline__ void load_a_frag(FragBF& f, const unsigned short* A, int lda, int lane) {
  int row = lane & 15, half = lane >> 4;
  const unsigned short* p = A + (size_t)row * lda;
#pragma unroll
  for (int v = 0; v < 8; ++v) {
    int k = (v < 4) ? (8 * half + 2 * v) : (16 + 8 * half + 2 * (v - 4));
    f.u[v] = *(const unsigned int*)(p + k);
  }
}

// B fragment: 32x16 bf16 from K-major storage Bt[N][K], pre-offset to (nbase,kbase).
// lane = column n, half selects K {0..7,16..23} / {8..15,24..31}, VGPR v packs a K pair -> dword load.
__device__ __forceinline__ void load_b_frag(FragBF& f, const unsigned short* Bt, int ldb, int lane) {
  int n = lane & 15, half = lane >> 4;
  const unsigned short* p = Bt + (size_t)n * ldb;
#pragma unroll
  for (int v = 0; v < 8; ++v) f.u[v] = *(const unsigned int*)(p + ((v < 4) ? (8 * half + 2 * v) : (16 + 8 * half + 2 * (v - 4))));
}

// CDNA5 async global->LDS copy (ASYNCcnt-tracked), GV addressing mode.
__device__ __forceinline__ void async_copy_b128(const unsigned short* gsrc, unsigned short* lds_dst) {
  unsigned loff = (unsigned)(size_t)lds_dst;   // low 32 bits of LDS aperture = LDS offset
  asm volatile("global_load_async_to_lds_b128 %0, %1, off"
               :: "v"(loff), "v"(gsrc)
               : "memory");
}
__device__ __forceinline__ void wait_async_le2() {
  asm volatile("s_wait_asynccnt 0x2" ::: "memory");
}

// ---------------------------------------------------------------------------
// Embedding gather: e[b,t,:] = tok_emb[x[b,t],:] + pos_emb[t,:]  (f32 + bf16)
// ---------------------------------------------------------------------------
__global__ __launch_bounds__(256) void embed_kernel(const int* __restrict__ x,
                                                    const float* __restrict__ tok,
                                                    const float* __restrict__ pos,
                                                    float* __restrict__ ef,
                                                    unsigned short* __restrict__ eb) {
  int row = blockIdx.x;            // 0..MTOK-1
  int t = row & (TT - 1);
  int id = x[row];
  const float* tp = tok + (size_t)id * EE;
  const float* pp = pos + (size_t)t * EE;
  float* eo = ef + (size_t)row * EE;
  unsigned short* bo = eb + (size_t)row * EE;
#pragma unroll
  for (int i = 0; i < 4; ++i) {
    int c = threadIdx.x + 256 * i;
    float v = tp[c] + pp[c];
    eo[c] = v;
    bo[c] = f2bf(v);
  }
}

// ---------------------------------------------------------------------------
// Weight convert + transpose: W[K][N] f32 -> Wt[N][K] bf16
// ---------------------------------------------------------------------------
__global__ __launch_bounds__(256) void cvtT_kernel(const float* __restrict__ W,
                                                   unsigned short* __restrict__ Wt,
                                                   int K, int N) {
  size_t idx = (size_t)blockIdx.x * 256 + threadIdx.x;
  int n = (int)(idx / K);
  int k = (int)(idx % K);
  Wt[idx] = f2bf(W[(size_t)k * N + n]);
}

// ---------------------------------------------------------------------------
// GEMM: C[M,N] = A[M,K](bf16) x Wt[N,K]^T (+bias, optional ReLU).
// 256 threads = 8 waves; block tile 64(M) x 128(N); wave computes 16x64.
// A tiles (64x64) are double-buffered in LDS via async global->LDS copies so the
// next tile's DMA overlaps this tile's WMMAs; both N-half wave groups share it.
// vt=1: write bf16 output transposed per head into [B][H][HD][T] (for V).
// ---------------------------------------------------------------------------
__global__ __launch_bounds__(256) void gemm_bf16_kernel(const unsigned short* __restrict__ A,
                                                        const unsigned short* __restrict__ Wt,
                                                        const float* __restrict__ bias,
                                                        float* __restrict__ Cf,
                                                        unsigned short* __restrict__ Cb,
                                                        int M, int N, int K, int relu, int vt) {
  __shared__ unsigned short ldsA[2][64 * 64];

  int tid = threadIdx.x;
  int lane = tid & 31;
  int wid  = tid >> 5;
  int wm = wid & 3, wn = wid >> 2;
  int rbase = blockIdx.y * 64 + wm * 16;
  int nbase = blockIdx.x * 128 + wn * 64;

  const unsigned short* Ablk = A + (size_t)(blockIdx.y * 64) * K;

  auto issue_tile = [&](int pb, int kb) {
#pragma unroll
    for (int i = 0; i < 2; ++i) {
      int c = tid + 256 * i;            // 512 x 16B chunks = 64 rows x 64 bf16
      int row = c >> 3;
      int cc  = (c & 7) * 8;
      async_copy_b128(Ablk + (size_t)row * K + kb + cc, &ldsA[pb][row * 64 + cc]);
    }
  };

  v8f acc[4];
#pragma unroll
  for (int j = 0; j < 4; ++j) acc[j] = zero_v8f();

  issue_tile(0, 0);

  int pb = 0;
  for (int kb = 0; kb < K; kb += 64, pb ^= 1) {
    int kb_next = (kb + 64 < K) ? (kb + 64) : kb;   // always-issue keeps wait imm constant
    issue_tile(pb ^ 1, kb_next);
    wait_async_le2();                                // previous tile's 2 asyncs done
    __syncthreads();
#pragma unroll
    for (int kk = 0; kk < 64; kk += 32) {
      FragBF a;
      load_a_frag(a, &ldsA[pb][wm * 16 * 64 + kk], 64, lane);
      FragBF b[4];
#pragma unroll
      for (int j = 0; j < 4; ++j)
        load_b_frag(b[j], Wt + (size_t)(nbase + 16 * j) * K + kb + kk, K, lane);
#pragma unroll
      for (int j = 0; j < 4; ++j)
        acc[j] = __builtin_amdgcn_wmma_f32_16x16x32_bf16(false, a.v, false, b[j].v,
                                                         (short)0, acc[j], false, false);
    }
    __syncthreads();   // all waves done reading before buffer pb is overwritten
  }

  int col = lane & 15, half = lane >> 4;
  if (vt) {
    // V path: no bias/relu/f32; write [B][H][HD][T]; 8 rows per lane are contiguous in t.
    int bidx = rbase >> 11;              // rbase / TT
    int t0 = (rbase & (TT - 1)) + 8 * half;
#pragma unroll
    for (int j = 0; j < 4; ++j) {
      int nc = nbase + 16 * j + col;
      int hh = nc >> 6, dd = nc & 63;
      unsigned short* dst = Cb + (((size_t)bidx * HH + hh) * HD + dd) * TT + t0;
#pragma unroll
      for (int r2 = 0; r2 < 4; ++r2) {
        unsigned int lo = f2bf(acc[j][2 * r2]);
        unsigned int hi = f2bf(acc[j][2 * r2 + 1]);
        *(unsigned int*)(dst + 2 * r2) = lo | (hi << 16);
      }
    }
    return;
  }
#pragma unroll
  for (int j = 0; j < 4; ++j) {
    int nc = nbase + 16 * j + col;
    float bv = bias ? bias[nc] : 0.0f;
#pragma unroll
    for (int r = 0; r < 8; ++r) {
      float xv = acc[j][r] + bv;
      if (relu) xv = fmaxf(xv, 0.0f);
      int rowg = rbase + r + 8 * half;
      if (Cf) Cf[(size_t)rowg * N + nc] = xv;
      if (Cb) Cb[(size_t)rowg * N + nc] = f2bf(xv);
    }
  }
}

// ---------------------------------------------------------------------------
// Flash attention: grid (T/64, B*H); 128 threads = 4 waves; each wave owns
// 16 q rows x full HD=64. Online softmax; P restaged via LDS into A-fragment
// form. Q/K are bf16 [B][T][E]; V is bf16 [B][H][HD][T] (K-major per head).
// ---------------------------------------------------------------------------
__global__ __launch_bounds__(128) void flash_attn_kernel(const unsigned short* __restrict__ Q,
                                                         const unsigned short* __restrict__ Km,
                                                         const unsigned short* __restrict__ Vt,
                                                         unsigned short* __restrict__ O,
                                                         int causal) {
  __shared__ unsigned short lds_p[4][16][72];   // per-wave P tile (16x64, padded)

  int lane = threadIdx.x & 31;
  int wid  = threadIdx.x >> 5;
  int bh = blockIdx.y;
  int b = bh >> 4, h = bh & 15;
  int qtile = blockIdx.x * 64;
  int qbase = qtile + wid * 16;
  size_t base = ((size_t)b * TT) * EE + (size_t)h * HD;
  const unsigned short* qp = Q + base;
  const unsigned short* kp = Km + base;
  const unsigned short* vp = Vt + (size_t)bh * HD * TT;   // [HD][T]

  int col = lane & 15, half = lane >> 4;

  v8f o[4];
  float m[8], l[8];
#pragma unroll
  for (int j = 0; j < 4; ++j) o[j] = zero_v8f();
#pragma unroll
  for (int r = 0; r < 8; ++r) { m[r] = -1.0e30f; l[r] = 0.0f; }

  int kend = causal ? (qtile + 64) : TT;

  for (int kt = 0; kt < kend; kt += 64) {
    // ---- S = Q @ K^T (16x64 score tile, HD=64 -> two 32-deep k-steps)
    v8f s[4];
#pragma unroll
    for (int j = 0; j < 4; ++j) s[j] = zero_v8f();
#pragma unroll
    for (int kk = 0; kk < 64; kk += 32) {
      FragBF qa;
      load_a_frag(qa, qp + (size_t)qbase * EE + kk, EE, lane);
      FragBF kf[4];
#pragma unroll
      for (int j = 0; j < 4; ++j)
        load_b_frag(kf[j], kp + (size_t)(kt + 16 * j) * EE + kk, EE, lane);
#pragma unroll
      for (int j = 0; j < 4; ++j)
        s[j] = __builtin_amdgcn_wmma_f32_16x16x32_bf16(false, qa.v, false, kf[j].v,
                                                       (short)0, s[j], false, false);
    }

    // ---- scale + causal mask
#pragma unroll
    for (int j = 0; j < 4; ++j) {
#pragma unroll
      for (int r = 0; r < 8; ++r) {
        float xv = s[j][r] * 0.125f;   // 1/sqrt(64)
        if (causal) {
          int cg = kt + 16 * j + col;
          int rg = qbase + r + 8 * half;
          if (cg > rg) xv = -1.0e30f;
        }
        s[j][r] = xv;
      }
    }

    // ---- online softmax row stats (row r+8*half lives in one 16-lane half)
    float corr[8];
#pragma unroll
    for (int r = 0; r < 8; ++r) {
      float t = fmaxf(fmaxf(s[0][r], s[1][r]), fmaxf(s[2][r], s[3][r]));
      t = fmaxf(t, __shfl_xor(t, 1));
      t = fmaxf(t, __shfl_xor(t, 2));
      t = fmaxf(t, __shfl_xor(t, 4));
      t = fmaxf(t, __shfl_xor(t, 8));
      float mn = fmaxf(m[r], t);
      corr[r] = __expf(m[r] - mn);
      m[r] = mn;
    }
    float rs[8];
#pragma unroll
    for (int r = 0; r < 8; ++r) rs[r] = 0.0f;
#pragma unroll
    for (int j = 0; j < 4; ++j) {
#pragma unroll
      for (int r = 0; r < 8; ++r) {
        float p = __expf(s[j][r] - m[r]);
        s[j][r] = p;
        rs[r] += p;
      }
    }
#pragma unroll
    for (int r = 0; r < 8; ++r) {
      float t = rs[r];
      t += __shfl_xor(t, 1);
      t += __shfl_xor(t, 2);
      t += __shfl_xor(t, 4);
      t += __shfl_xor(t, 8);
      l[r] = l[r] * corr[r] + t;
    }
#pragma unroll
    for (int j = 0; j < 4; ++j)
#pragma unroll
      for (int r = 0; r < 8; ++r) o[j][r] *= corr[r];

    // ---- restage P (C-layout) through LDS into A-fragment layout
#pragma unroll
    for (int j = 0; j < 4; ++j)
#pragma unroll
      for (int r = 0; r < 8; ++r)
        lds_p[wid][r + 8 * half][16 * j + col] = f2bf(s[j][r]);
    // per-wave producer/consumer; compiler inserts s_wait_dscnt

    // ---- O += P @ V  (V is K-major [HD][T]: dword-pair B fragments)
#pragma unroll
    for (int kk = 0; kk < 64; kk += 32) {
      FragBF pa;
      load_a_frag(pa, &lds_p[wid][0][kk], 72, lane);
      FragBF vf[4];
#pragma unroll
      for (int j = 0; j < 4; ++j)
        load_b_frag(vf[j], vp + (size_t)(16 * j) * TT + kt + kk, TT, lane);
#pragma unroll
      for (int j = 0; j < 4; ++j)
        o[j] = __builtin_amdgcn_wmma_f32_16x16x32_bf16(false, pa.v, false, vf[j].v,
                                                       (short)0, o[j], false, false);
    }
  }

  // ---- normalize and write bf16 output (feeds the Wo GEMM)
#pragma unroll
  for (int j = 0; j < 4; ++j) {
#pragma unroll
    for (int r = 0; r < 8; ++r) {
      int rg = qbase + r + 8 * half;
      float val = o[j][r] / l[r];
      O[((size_t)b * TT + rg) * EE + h * HD + 16 * j + col] = f2bf(val);
    }
  }
}

// ---------------------------------------------------------------------------
// Residual add + LayerNorm: out = LN(a + b) * g + beta ; f32 + optional bf16
// ---------------------------------------------------------------------------
__global__ __launch_bounds__(256) void ln_kernel(const float* __restrict__ a,
                                                 const float* __restrict__ b,
                                                 const float* __restrict__ g,
                                                 const float* __restrict__ beta,
                                                 float* __restrict__ outf,
                                                 unsigned short* __restrict__ outb) {
  __shared__ float red[256];
  int row = blockIdx.x;
  int tid = threadIdx.x;
  const float* ap = a + (size_t)row * EE;
  const float* bp = b + (size_t)row * EE;
  float xv[4];
  float s = 0.0f, ss = 0.0f;
#pragma unroll
  for (int i = 0; i < 4; ++i) {
    int c = tid + 256 * i;
    float v = ap[c] + bp[c];
    xv[i] = v;
    s += v;
    ss += v * v;
  }
  red[tid] = s;
  __syncthreads();
  for (int off = 128; off > 0; off >>= 1) {
    if (tid < off) red[tid] += red[tid + off];
    __syncthreads();
  }
  float mean = red[0] * (1.0f / EE);
  __syncthreads();
  red[tid] = ss;
  __syncthreads();
  for (int off = 128; off > 0; off >>= 1) {
    if (tid < off) red[tid] += red[tid + off];
    __syncthreads();
  }
  float var = red[0] * (1.0f / EE) - mean * mean;
  float inv = rsqrtf(var + 1e-5f);
#pragma unroll
  for (int i = 0; i < 4; ++i) {
    int c = tid + 256 * i;
    float v = (xv[i] - mean) * inv * g[c] + beta[c];
    if (outf) outf[(size_t)row * EE + c] = v;
    if (outb) outb[(size_t)row * EE + c] = f2bf(v);
  }
}

// ---------------------------------------------------------------------------
// Launch
// ---------------------------------------------------------------------------
extern "C" void kernel_launch(void* const* d_in, const int* in_sizes, int n_in,
                              void* d_out, int out_size, void* d_ws, size_t ws_size,
                              hipStream_t stream) {
  const int*   x       = (const int*)d_in[0];
  const float* tok_emb = (const float*)d_in[1];
  const float* pos_emb = (const float*)d_in[2];
  const float* m_Wq = (const float*)d_in[3];
  const float* m_Wk = (const float*)d_in[4];
  const float* m_Wv = (const float*)d_in[5];
  const float* m_Wo = (const float*)d_in[6];
  const float* m_bo = (const float*)d_in[7];
  const float* h_Wq = (const float*)d_in[8];
  const float* h_Wk = (const float*)d_in[9];
  const float* h_Wv = (const float*)d_in[10];
  const float* h_Wo = (const float*)d_in[11];
  const float* h_bo = (const float*)d_in[12];
  const float* f_W1 = (const float*)d_in[13];
  const float* f_b1 = (const float*)d_in[14];
  const float* f_W2 = (const float*)d_in[15];
  const float* f_b2 = (const float*)d_in[16];
  const float* f_W3 = (const float*)d_in[17];
  const float* f_b3 = (const float*)d_in[18];
  const float* f_W4 = (const float*)d_in[19];
  const float* f_b4 = (const float*)d_in[20];
  const float* ln_g = (const float*)d_in[21];
  const float* ln_b = (const float*)d_in[22];
  float* out = (float*)d_out;

  char* ws = (char*)d_ws;
  size_t off = 0;
  auto alloc = [&](size_t bytes) { size_t o = off; off += (bytes + 255) & ~(size_t)255; return o; };

  const size_t f32buf = (size_t)MTOK * EE * 4;
  const size_t bf16buf = (size_t)MTOK * EE * 2;

  size_t oA   = alloc(f32buf);    // e, later reused for r2
  size_t oP   = alloc(f32buf);    // projection / f4 scratch
  size_t oR1  = alloc(f32buf);    // r1
  size_t oACT = alloc(bf16buf);   // bf16 activation (e -> r1 -> r2)
  size_t oQ   = alloc(bf16buf);   // q (later h1)
  size_t oK   = alloc(bf16buf);   // k (later h2)
  size_t oV   = alloc(bf16buf);   // v^T per head (later h3)
  size_t oATT = alloc(bf16buf);   // attention output

  size_t oWT[12];
  const float* Wsrc[12] = {m_Wq, m_Wk, m_Wv, m_Wo, h_Wq, h_Wk, h_Wv, h_Wo, f_W1, f_W2, f_W3, f_W4};
  const int WK[12] = {EE, EE, EE, EE, EE, EE, EE, EE, EE, 512, EE, 512};
  const int WN[12] = {EE, EE, EE, EE, EE, EE, EE, EE, 512, EE, 512, EE};
  for (int i = 0; i < 12; ++i) oWT[i] = alloc((size_t)WK[i] * WN[i] * 2);
  (void)ws_size; (void)n_in; (void)in_sizes; (void)out_size;

  float*          Af   = (float*)(ws + oA);
  float*          Pf   = (float*)(ws + oP);
  float*          R1f  = (float*)(ws + oR1);
  unsigned short* ACT  = (unsigned short*)(ws + oACT);
  unsigned short* Qb   = (unsigned short*)(ws + oQ);
  unsigned short* Kb   = (unsigned short*)(ws + oK);
  unsigned short* Vb   = (unsigned short*)(ws + oV);
  unsigned short* ATT  = (unsigned short*)(ws + oATT);
  unsigned short* WT[12];
  for (int i = 0; i < 12; ++i) WT[i] = (unsigned short*)(ws + oWT[i]);

  // 0) weight convert+transpose
  for (int i = 0; i < 12; ++i) {
    size_t total = (size_t)WK[i] * WN[i];
    cvtT_kernel<<<dim3((unsigned)(total / 256)), dim3(256), 0, stream>>>(Wsrc[i], WT[i], WK[i], WN[i]);
  }

  // 1) embeddings
  embed_kernel<<<dim3(MTOK), dim3(256), 0, stream>>>(x, tok_emb, pos_emb, Af, ACT);

  auto gemm = [&](const unsigned short* A, const unsigned short* Wt, const float* bias,
                  float* Cf, unsigned short* Cb, int N, int K, int relu, int vt) {
    dim3 grid((unsigned)(N / 128), (unsigned)(MTOK / 64));
    gemm_bf16_kernel<<<grid, dim3(256), 0, stream>>>(A, Wt, bias, Cf, Cb, MTOK, N, K, relu, vt);
  };
  dim3 agrid((unsigned)(TT / 64), (unsigned)(BB * HH));

  // ---- Block 1: masked MHA ----
  gemm(ACT, WT[0], nullptr, nullptr, Qb, EE, EE, 0, 0);
  gemm(ACT, WT[1], nullptr, nullptr, Kb, EE, EE, 0, 0);
  gemm(ACT, WT[2], nullptr, nullptr, Vb, EE, EE, 0, 1);   // V in [B][H][HD][T]
  flash_attn_kernel<<<agrid, dim3(128), 0, stream>>>(Qb, Kb, Vb, ATT, 1);
  gemm(ATT, WT[3], m_bo, Pf, nullptr, EE, EE, 0, 0);
  ln_kernel<<<dim3(MTOK), dim3(256), 0, stream>>>(Af, Pf, ln_g, ln_b, R1f, ACT);

  // ---- Block 2: unmasked MHA ----
  gemm(ACT, WT[4], nullptr, nullptr, Qb, EE, EE, 0, 0);
  gemm(ACT, WT[5], nullptr, nullptr, Kb, EE, EE, 0, 0);
  gemm(ACT, WT[6], nullptr, nullptr, Vb, EE, EE, 0, 1);
  flash_attn_kernel<<<agrid, dim3(128), 0, stream>>>(Qb, Kb, Vb, ATT, 0);
  gemm(ATT, WT[7], h_bo, Pf, nullptr, EE, EE, 0, 0);
  ln_kernel<<<dim3(MTOK), dim3(256), 0, stream>>>(R1f, Pf, ln_g, ln_b, Af, ACT);  // r2 -> Af/ACT

  // ---- Block 3: FFN ----
  gemm(ACT, WT[8],  f_b1, nullptr, Qb, 512, EE,  1, 0);   // h1 (relu)
  gemm(Qb,  WT[9],  f_b2, nullptr, Kb, EE,  512, 1, 0);   // h2 (relu)
  gemm(Kb,  WT[10], f_b3, nullptr, Vb, 512, EE,  1, 0);   // h3 (relu)
  gemm(Vb,  WT[11], f_b4, Pf, nullptr, EE, 512, 0, 0);    // f4
  ln_kernel<<<dim3(MTOK), dim3(256), 0, stream>>>(Af, Pf, ln_g, ln_b, out, nullptr);
}
